// Qwen3Attention_72602127171975
// MI455X (gfx1250) — matrix operand order repaired and checked by reading, not compiled, not run on hardware
//
#include <hip/hip_runtime.h>

#define HIDDEN   4096
#define N_HEADS  32
#define N_KV     8
#define HEAD_DIM 128
#define BATCH    4
#define SEQ      1024
#define TOK      (BATCH*SEQ)
#define QKV_OUT  ((N_HEADS + 2*N_KV)*HEAD_DIM)   // 6144
#define EPSF     1e-6f
#define THETA    10000.0f
#define SCALEF   0.08838834764831845f            // 128^-0.5

typedef __bf16 bf16;
typedef __attribute__((ext_vector_type(16))) __bf16 v16bf;
typedef __attribute__((ext_vector_type(8)))  __bf16 v8bf;
typedef __attribute__((ext_vector_type(8)))  float  v8f;
typedef __attribute__((ext_vector_type(4)))  unsigned int v4u;
typedef __attribute__((ext_vector_type(8)))  int v8i;
typedef __attribute__((ext_vector_type(4)))  int v4i;

static __device__ __forceinline__ v16bf cat8(v8bf lo, v8bf hi) {
  return __builtin_shufflevector(lo, hi, 0,1,2,3,4,5,6,7,8,9,10,11,12,13,14,15);
}
static __device__ __forceinline__ v8f vzero8() {
  v8f z;
#pragma unroll
  for (int i = 0; i < 8; ++i) z[i] = 0.0f;
  return z;
}
static __device__ __forceinline__ v8f wmma_bf16(v16bf a, v16bf b, v8f c) {
  // D = A(16x32 bf16) x B(32x16 bf16) + C(16x16 f32)
  return __builtin_amdgcn_wmma_f32_16x16x32_bf16(false, a, false, b, (short)0, c,
                                                 false, false);
}
// A-fragment (16x32): lane L<16 -> row L, K = {0..7, 16..23}; lanes 16..31 -> K = {8..15, 24..31}
static __device__ __forceinline__ v16bf afrag_lds(const bf16* base, int stride,
                                                  int row0, int kbase, int lane) {
  const bf16* p = base + (size_t)(row0 + (lane & 15)) * stride + kbase + ((lane >> 4) << 3);
  return cat8(*(const v8bf*)p, *(const v8bf*)(p + 16));
}
// B-fragment (32x16): lane L -> col (L&15), K = {0..7, 16..23} (L<16) / {8..15, 24..31}; storage is n-major, K contiguous
static __device__ __forceinline__ v16bf bfrag_lds(const bf16* base, int stride,
                                                  int row0, int kbase, int lane) {
  const bf16* p = base + (size_t)(row0 + (lane & 15)) * stride + kbase + ((lane >> 4) << 3);
  return cat8(*(const v8bf*)p, *(const v8bf*)(p + 16));
}

// ---------------------------------------------------------------------------
// Tensor Data Mover: DMA a 2-D tile (128 rows x 64 bytes, global row stride
// `stride_bytes`) into LDS at byte offset `lds_off`, padding each 64B row by
// 32B (pad_interval=16 DWORDs, pad_amount=8 DWORDs) -> 96B LDS rows, which is
// exactly the bf16[128][48] layout the WMMA fragment readers use.
// D# bitfields per CDNA5 ISA (count=1, type=2, 1-byte
// data_size, groups 2/3 zero for a 2-D tensor). Tracked by TENSORcnt.
// This toolchain exposes the 6-arg builtin: (v4u, v8i, v4i, v4i, v8i, cpol).
static __device__ __forceinline__ void tdm_load_tile(const void* g, unsigned lds_off,
                                                     unsigned stride_bytes) {
  const unsigned long long ga = (unsigned long long)(uintptr_t)g;
  v4u g0;
  g0[0] = 1u;                                               // count=1
  g0[1] = lds_off;                                          // lds_addr
  g0[2] = (unsigned)ga;                                     // global_addr[31:0]
  g0[3] = (unsigned)((ga >> 32) & 0x01FFFFFFu) | 0x80000000u; // [56:32] | type=2
  v8i g1;
  g1[0] = (int)((1u << 20) | (3u << 22) | (7u << 25));      // pad_en, interval=16DW, amount=8DW
  g1[1] = (int)(64u << 16);                                 // tensor_dim0 = 64 bytes
  g1[2] = (int)(128u << 16);                                // tensor_dim1 = 128 rows
  g1[3] = (int)(64u << 16);                                 // tile_dim0 = 64 bytes
  g1[4] = (int)128u;                                        // tile_dim1 = 128, tile_dim2 = 0
  g1[5] = (int)stride_bytes;                                // tensor_dim0_stride[31:0]
  g1[6] = 0;
  g1[7] = 0;
  v4i gz4;
#pragma unroll
  for (int i = 0; i < 4; ++i) gz4[i] = 0;
  v8i gz8;
#pragma unroll
  for (int i = 0; i < 8; ++i) gz8[i] = 0;
  __builtin_amdgcn_tensor_load_to_lds(g0, g1, gz4, gz4, gz8, 0);
}

// ---------------------------------------------------------------------------
// f32 -> bf16 row-major convert
__global__ __launch_bounds__(256) void cvt_bf16_kernel(const float* __restrict__ in,
                                                       bf16* __restrict__ out, size_t n) {
  for (size_t i = (size_t)blockIdx.x * blockDim.x + threadIdx.x; i < n;
       i += (size_t)gridDim.x * blockDim.x)
    out[i] = (bf16)in[i];
}

// f32[R][C] -> bf16[C][R] (tiled transpose; dims are multiples of 32)
__global__ __launch_bounds__(256) void transpose_bf16_kernel(const float* __restrict__ in,
                                                             bf16* __restrict__ out,
                                                             int R, int Ccols) {
  __shared__ float tile[32][33];
  const int c0 = blockIdx.x * 32, r0 = blockIdx.y * 32;
  const int tx = threadIdx.x, ty = threadIdx.y;   // block (32, 8)
#pragma unroll
  for (int i = 0; i < 32; i += 8)
    tile[ty + i][tx] = in[(size_t)(r0 + ty + i) * Ccols + c0 + tx];
  __syncthreads();
#pragma unroll
  for (int i = 0; i < 32; i += 8)
    out[(size_t)(c0 + ty + i) * R + r0 + tx] = (bf16)tile[tx][ty + i];
}

// ---------------------------------------------------------------------------
// C[M,N](f32) = A[M,K](bf16, row-major) x Bt[N,K](bf16)   (optionally * (*scale))
// Block: 256 thr = 8 waves; block tile 128x128, wave tile 64x32, K-step 32.
// Double-buffered LDS; tiles staged by the Tensor Data Mover from wave 0 while
// all waves run WMMAs on the other buffer; one barrier per K-step.
__global__ __launch_bounds__(256) void gemm_bf16_kernel(const bf16* __restrict__ A,
                                                        const bf16* __restrict__ Bt,
                                                        float* __restrict__ C,
                                                        int M, int N, int K,
                                                        const float* __restrict__ scale_ptr) {
  __shared__ bf16 As[2][128][48];   // 96B row stride: 16B aligned, bank-staggered
  __shared__ bf16 Bs[2][128][48];
  const int tid  = threadIdx.x;
  const int lane = tid & 31;
  const int wave = tid >> 5;
  const int wy = wave >> 2;      // 0..1 -> 64-row half
  const int wx = wave & 3;       // 0..3 -> 32-col quarter
  const int bm = blockIdx.y * 128;
  const int bn = blockIdx.x * 128;

  v8f acc[4][2];
#pragma unroll
  for (int i = 0; i < 4; ++i)
#pragma unroll
    for (int j = 0; j < 2; ++j) acc[i][j] = vzero8();

  const unsigned ldsA0 = (unsigned)(uintptr_t)&As[0][0][0];
  const unsigned ldsA1 = (unsigned)(uintptr_t)&As[1][0][0];
  const unsigned ldsB0 = (unsigned)(uintptr_t)&Bs[0][0][0];
  const unsigned ldsB1 = (unsigned)(uintptr_t)&Bs[1][0][0];
  const char* gA = (const char*)(A  + (size_t)bm * K);
  const char* gB = (const char*)(Bt + (size_t)bn * K);
  const unsigned stride = (unsigned)K * 2u;   // global row stride in bytes

  if (wave == 0) {
    tdm_load_tile(gA, ldsA0, stride);
    tdm_load_tile(gB, ldsB0, stride);
    __builtin_amdgcn_s_wait_tensorcnt(0);
  }
  __syncthreads();

  const int nk = K / 32;
  for (int kt = 0; kt < nk; ++kt) {
    const int cur = kt & 1;
    if (wave == 0 && kt + 1 < nk) {   // prefetch next K-tile via TDM into other buffer
      const size_t off = (size_t)(kt + 1) * 64;   // 32 bf16 = 64 bytes
      tdm_load_tile(gA + off, cur ? ldsA0 : ldsA1, stride);
      tdm_load_tile(gB + off, cur ? ldsB0 : ldsB1, stride);
    }
    v16bf af[4];
#pragma unroll
    for (int i = 0; i < 4; ++i)
      af[i] = afrag_lds(&As[cur][0][0], 48, wy * 64 + i * 16, 0, lane);
#pragma unroll
    for (int j = 0; j < 2; ++j) {
      v16bf bf_ = bfrag_lds(&Bs[cur][0][0], 48, wx * 32 + j * 16, 0, lane);
#pragma unroll
      for (int i = 0; i < 4; ++i) acc[i][j] = wmma_bf16(af[i], bf_, acc[i][j]);
    }
    if (wave == 0) __builtin_amdgcn_s_wait_tensorcnt(0);
    __syncthreads();
  }

  const float s = scale_ptr ? *scale_ptr : 1.0f;
#pragma unroll
  for (int i = 0; i < 4; ++i)
#pragma unroll
    for (int j = 0; j < 2; ++j)
#pragma unroll
      for (int r = 0; r < 8; ++r) {
        const int row = bm + wy * 64 + i * 16 + r + ((lane >> 4) << 3);
        const int col = bn + wx * 32 + j * 16 + (lane & 15);
        C[(size_t)row * N + col] = acc[i][j][r] * s;
      }
}

// ---------------------------------------------------------------------------
// One wave per (token, head-slot). slots: 0..31 = Q heads, 32..39 = K heads, 40..47 = V heads.
// RMSNorm (Q/K) + RoPE (Q/K), bf16 outputs; also saves Q inv-rms for router pooling (pre-RoPE).
__global__ __launch_bounds__(32) void normrope_kernel(const float* __restrict__ qkv,
                                                      const int* __restrict__ positions,
                                                      const float* __restrict__ qw,
                                                      const float* __restrict__ kw,
                                                      bf16* __restrict__ qb,
                                                      bf16* __restrict__ kb,
                                                      bf16* __restrict__ vb,
                                                      float* __restrict__ inv_rms_q) {
  const int blk  = blockIdx.x;
  const int t    = blk / 48;
  const int r    = blk % 48;
  const int lane = threadIdx.x;
  const float* x = qkv + (size_t)t * QKV_OUT + r * HEAD_DIM;
  const float x0 = x[lane], x1 = x[lane + 32], x2 = x[lane + 64], x3 = x[lane + 96];

  if (r >= 40) {   // V: plain bf16 convert
    bf16* o = vb + ((size_t)t * N_KV + (r - 40)) * HEAD_DIM;
    o[lane] = (bf16)x0; o[lane + 32] = (bf16)x1; o[lane + 64] = (bf16)x2; o[lane + 96] = (bf16)x3;
    return;
  }
  const bool isq = (r < 32);
  const float* w = isq ? qw : kw;
  float ss = x0 * x0 + x1 * x1 + x2 * x2 + x3 * x3;
#pragma unroll
  for (int m = 16; m >= 1; m >>= 1) ss += __shfl_xor(ss, m, 32);
  const float inv = rsqrtf(ss * (1.0f / 128.0f) + EPSF);
  if (isq && lane == 0) inv_rms_q[t * N_HEADS + r] = inv;

  const float n0 = x0 * inv * w[lane];
  const float n1 = x1 * inv * w[lane + 32];
  const float n2 = x2 * inv * w[lane + 64];
  const float n3 = x3 * inv * w[lane + 96];
  const float pos = (float)positions[t];
  const float f0 = pos * powf(THETA, -(float)lane / 64.0f);
  const float f1 = pos * powf(THETA, -(float)(lane + 32) / 64.0f);
  const float c0 = cosf(f0), s0 = sinf(f0), c1 = cosf(f1), s1 = sinf(f1);
  const float o0 = n0 * c0 - n2 * s0, o2 = n2 * c0 + n0 * s0;
  const float o1 = n1 * c1 - n3 * s1, o3 = n3 * c1 + n1 * s1;
  bf16* o = isq ? qb + ((size_t)t * N_HEADS + r) * HEAD_DIM
                : kb + ((size_t)t * N_KV + (r - 32)) * HEAD_DIM;
  o[lane] = (bf16)o0; o[lane + 32] = (bf16)o1; o[lane + 64] = (bf16)o2; o[lane + 96] = (bf16)o3;
}

// ---------------------------------------------------------------------------
// pooled[b][d] = mean over (s,h) of rmsnorm(q)[pre-RoPE] = qw[d] * mean(q_raw * inv_rms)
__global__ __launch_bounds__(256) void pool_kernel(const float* __restrict__ qkv,
                                                   const float* __restrict__ inv_rms_q,
                                                   const float* __restrict__ qw,
                                                   float* __restrict__ pooled) {
  const int b = blockIdx.x >> 7;
  const int d = blockIdx.x & 127;
  const int tid = threadIdx.x;
  float sum = 0.0f;
  for (int i = tid; i < SEQ * N_HEADS; i += 256) {
    const int s = i >> 5, h = i & 31;
    const int t = b * SEQ + s;
    sum += qkv[(size_t)t * QKV_OUT + h * HEAD_DIM + d] * inv_rms_q[t * N_HEADS + h];
  }
  __shared__ float red[256];
  red[tid] = sum;
  __syncthreads();
#pragma unroll
  for (int m = 128; m >= 1; m >>= 1) {
    if (tid < m) red[tid] += red[tid + m];
    __syncthreads();
  }
  if (tid == 0) pooled[b * HEAD_DIM + d] = red[0] * qw[d] * (1.0f / (SEQ * N_HEADS));
}

// Tiny router MLP -> scalar gate
__global__ __launch_bounds__(256) void router_kernel(const float* __restrict__ pooled,
    const float* w1, const float* b1, const float* w2, const float* b2,
    const float* w3, const float* b3, const float* w4, const float* b4,
    const float* w5, const float* b5, float* __restrict__ gate) {
  __shared__ float p[4 * 128];
  __shared__ float h1[4 * 1024];
  __shared__ float h2[4 * 256];
  __shared__ float h3[4 * 512];
  __shared__ float h4[4 * 128];
  const int tid = threadIdx.x;
  for (int i = tid; i < 4 * 128; i += 256) p[i] = pooled[i];
  __syncthreads();
  for (int i = tid; i < 4 * 1024; i += 256) {
    const int b = i >> 10, o = i & 1023;
    float acc = b1[o];
    for (int k = 0; k < 128; ++k) acc += p[b * 128 + k] * w1[k * 1024 + o];
    h1[i] = acc / (1.0f + expf(-acc));
  }
  __syncthreads();
  for (int i = tid; i < 4 * 256; i += 256) {
    const int b = i >> 8, o = i & 255;
    float acc = b2[o];
    for (int k = 0; k < 1024; ++k) acc += h1[b * 1024 + k] * w2[k * 256 + o];
    h2[i] = acc;
  }
  __syncthreads();
  for (int i = tid; i < 4 * 512; i += 256) {
    const int b = i >> 9, o = i & 511;
    float acc = b3[o];
    for (int k = 0; k < 256; ++k) acc += h2[b * 256 + k] * w3[k * 512 + o];
    h3[i] = acc / (1.0f + expf(-acc));
  }
  __syncthreads();
  for (int i = tid; i < 4 * 128; i += 256) {
    const int b = i >> 7, o = i & 127;
    float acc = b4[o];
    for (int k = 0; k < 512; ++k) acc += h3[b * 512 + k] * w4[k * 128 + o];
    h4[i] = acc / (1.0f + expf(-acc));
  }
  __syncthreads();
  if (tid == 0) {
    float any = 0.0f;
    for (int b = 0; b < 4; ++b) {
      float l0 = b5[0], l1 = b5[1];
      for (int k = 0; k < 128; ++k) {
        l0 += h4[b * 128 + k] * w5[k * 2];
        l1 += h4[b * 128 + k] * w5[k * 2 + 1];
      }
      if (l1 > l0) any = 1.0f;   // argmax==1 (ties -> 0)
    }
    *gate = any;
  }
}

// ---------------------------------------------------------------------------
// Flash attention: one wave per (16-query tile, b*h). 32-key tiles, online softmax.
__global__ __launch_bounds__(32) void flash_kernel(const bf16* __restrict__ qb,
                                                   const bf16* __restrict__ kb,
                                                   const bf16* __restrict__ vb,
                                                   bf16* __restrict__ ob) {
  __shared__ bf16 Ks[32][136];    // keys x d        (272B rows)
  __shared__ bf16 Vts[128][48];   // d x keys (transposed, 96B rows)
  __shared__ bf16 Ps[16][48];     // P tile, A-readable
  const int lane  = threadIdx.x;
  const int qt    = blockIdx.x;           // 0..63
  const int bh    = blockIdx.y;           // 0..127
  const int b     = bh >> 5, h = bh & 31;
  const int kvh   = h >> 2;               // GQA group of 4
  const int qbase = qt * 16;

  // Q A-fragments straight from global (row-major per lane)
  v16bf qf[4];
  {
    const int row = qbase + (lane & 15);
    const bf16* qp = qb + ((size_t)(b * SEQ + row) * N_HEADS + h) * HEAD_DIM;
    const int klo = (lane >> 4) << 3;
#pragma unroll
    for (int c = 0; c < 4; ++c)
      qf[c] = cat8(*(const v8bf*)(qp + c * 32 + klo),
                   *(const v8bf*)(qp + c * 32 + klo + 16));
  }

  v8f o[8];
#pragma unroll
  for (int nt = 0; nt < 8; ++nt) o[nt] = vzero8();
  float mrow[8], lrow[8];
#pragma unroll
  for (int r = 0; r < 8; ++r) { mrow[r] = -1e30f; lrow[r] = 0.0f; }

  const int nkt = (qbase + 15) / 32 + 1;
  for (int kt = 0; kt < nkt; ++kt) {
    const int kb0 = kt * 32;
    {  // stage K rows (row-major) and V rows (transposed) — one key row per lane
      const bf16* kp = kb + ((size_t)(b * SEQ + kb0 + lane) * N_KV + kvh) * HEAD_DIM;
      const bf16* vp = vb + ((size_t)(b * SEQ + kb0 + lane) * N_KV + kvh) * HEAD_DIM;
#pragma unroll
      for (int i = 0; i < 16; ++i)
        *(v8bf*)&Ks[lane][i * 8] = *(const v8bf*)(kp + i * 8);
#pragma unroll
      for (int i = 0; i < 16; ++i) {
        v8bf u = *(const v8bf*)(vp + i * 8);
#pragma unroll
        for (int j = 0; j < 8; ++j) Vts[i * 8 + j][lane] = u[j];
      }
      if (kt + 1 < nkt) {   // hint next key/value tile into cache (global_prefetch_b8)
        __builtin_prefetch(kp + (size_t)32 * N_KV * HEAD_DIM, 0, 1);
        __builtin_prefetch(vp + (size_t)32 * N_KV * HEAD_DIM, 0, 1);
      }
    }
    __syncthreads();

    // scores: 16x32 = two 16x16 C tiles, 4 d-chunks each
    v8f c0 = vzero8(), c1 = vzero8();
#pragma unroll
    for (int c = 0; c < 4; ++c) {
      c0 = wmma_bf16(qf[c], bfrag_lds(&Ks[0][0], 136, 0,  c * 32, lane), c0);
      c1 = wmma_bf16(qf[c], bfrag_lds(&Ks[0][0], 136, 16, c * 32, lane), c1);
    }

    // online softmax per row (cross-lane reductions within 16-lane halves)
    float alpha[8];
#pragma unroll
    for (int r = 0; r < 8; ++r) {
      const int qi  = qbase + r + ((lane >> 4) << 3);
      const int k0i = kb0 + (lane & 15);
      float s0 = c0[r] * SCALEF; if (k0i      > qi) s0 = -1e30f;
      float s1 = c1[r] * SCALEF; if (k0i + 16 > qi) s1 = -1e30f;
      float mx = fmaxf(s0, s1);
#pragma unroll
      for (int m = 8; m >= 1; m >>= 1) mx = fmaxf(mx, __shfl_xor(mx, m, 16));
      const float mnew = fmaxf(mrow[r], mx);
      alpha[r] = expf(mrow[r] - mnew);
      mrow[r]  = mnew;
      const float p0 = expf(s0 - mnew), p1 = expf(s1 - mnew);
      float rs = p0 + p1;
#pragma unroll
      for (int m = 8; m >= 1; m >>= 1) rs += __shfl_xor(rs, m, 16);
      lrow[r] = lrow[r] * alpha[r] + rs;
      const int mi = r + ((lane >> 4) << 3);
      Ps[mi][lane & 15]        = (bf16)p0;
      Ps[mi][(lane & 15) + 16] = (bf16)p1;
    }
    __syncthreads();

#pragma unroll
    for (int nt = 0; nt < 8; ++nt)
#pragma unroll
      for (int r = 0; r < 8; ++r) o[nt][r] *= alpha[r];

    const v16bf pf = afrag_lds(&Ps[0][0], 48, 0, 0, lane);
#pragma unroll
    for (int nt = 0; nt < 8; ++nt)
      o[nt] = wmma_bf16(pf, bfrag_lds(&Vts[0][0], 48, nt * 16, 0, lane), o[nt]);
    __syncthreads();
  }

#pragma unroll
  for (int nt = 0; nt < 8; ++nt)
#pragma unroll
    for (int r = 0; r < 8; ++r) {
      const int m   = r + ((lane >> 4) << 3);
      const int row = b * SEQ + qbase + m;
      const int col = h * HEAD_DIM + nt * 16 + (lane & 15);
      ob[(size_t)row * (N_HEADS * HEAD_DIM) + col] = (bf16)(o[nt][r] / lrow[r]);
    }
}

// ---------------------------------------------------------------------------
extern "C" void kernel_launch(void* const* d_in, const int* in_sizes, int n_in,
                              void* d_out, int out_size, void* d_ws, size_t ws_size,
                              hipStream_t stream) {
  const float* hidden    = (const float*)d_in[0];
  const int*   positions = (const int*)d_in[1];
  // d_in[2] cu_seqlens: layout is dense B x S, unused
  const float* qkv_w     = (const float*)d_in[3];
  const float* o_w       = (const float*)d_in[4];
  const float* q_norm_w  = (const float*)d_in[5];
  const float* k_norm_w  = (const float*)d_in[6];
  const float* r_w1 = (const float*)d_in[7];  const float* r_b1 = (const float*)d_in[8];
  const float* r_w2 = (const float*)d_in[9];  const float* r_b2 = (const float*)d_in[10];
  const float* r_w3 = (const float*)d_in[11]; const float* r_b3 = (const float*)d_in[12];
  const float* r_w4 = (const float*)d_in[13]; const float* r_b4 = (const float*)d_in[14];
  const float* r_w5 = (const float*)d_in[15]; const float* r_b5 = (const float*)d_in[16];
  float* out = (float*)d_out;

  char* w = (char*)d_ws;
  auto alloc = [&](size_t bytes) -> void* {
    void* p = (void*)w;
    w += (bytes + 255) & ~(size_t)255;
    return p;
  };
  bf16*  A_bf  = (bf16*) alloc((size_t)TOK * HIDDEN * 2);
  bf16*  Wqkv  = (bf16*) alloc((size_t)QKV_OUT * HIDDEN * 2);
  bf16*  Wo    = (bf16*) alloc((size_t)HIDDEN * HIDDEN * 2);
  float* qkvf  = (float*)alloc((size_t)TOK * QKV_OUT * 4);
  bf16*  qbuf  = (bf16*) alloc((size_t)TOK * N_HEADS * HEAD_DIM * 2);
  bf16*  kbuf  = (bf16*) alloc((size_t)TOK * N_KV * HEAD_DIM * 2);
  bf16*  vbuf  = (bf16*) alloc((size_t)TOK * N_KV * HEAD_DIM * 2);
  bf16*  attn  = (bf16*) alloc((size_t)TOK * N_HEADS * HEAD_DIM * 2);
  float* invr  = (float*)alloc((size_t)TOK * N_HEADS * 4);
  float* pooled= (float*)alloc(4 * 128 * 4);
  float* gate  = (float*)alloc(4);

  cvt_bf16_kernel<<<4096, 256, 0, stream>>>(hidden, A_bf, (size_t)TOK * HIDDEN);
  transpose_bf16_kernel<<<dim3(QKV_OUT / 32, HIDDEN / 32), dim3(32, 8), 0, stream>>>(
      qkv_w, Wqkv, HIDDEN, QKV_OUT);
  transpose_bf16_kernel<<<dim3(HIDDEN / 32, HIDDEN / 32), dim3(32, 8), 0, stream>>>(
      o_w, Wo, HIDDEN, HIDDEN);

  gemm_bf16_kernel<<<dim3(QKV_OUT / 128, TOK / 128), 256, 0, stream>>>(
      A_bf, Wqkv, qkvf, TOK, QKV_OUT, HIDDEN, nullptr);

  normrope_kernel<<<TOK * 48, 32, 0, stream>>>(qkvf, positions, q_norm_w, k_norm_w,
                                               qbuf, kbuf, vbuf, invr);
  pool_kernel<<<BATCH * HEAD_DIM, 256, 0, stream>>>(qkvf, invr, q_norm_w, pooled);
  router_kernel<<<1, 256, 0, stream>>>(pooled, r_w1, r_b1, r_w2, r_b2, r_w3, r_b3,
                                       r_w4, r_b4, r_w5, r_b5, gate);

  flash_kernel<<<dim3(SEQ / 16, BATCH * N_HEADS), 32, 0, stream>>>(qbuf, kbuf, vbuf, attn);

  gemm_bf16_kernel<<<dim3(HIDDEN / 128, TOK / 128), 256, 0, stream>>>(
      attn, Wo, out, TOK, HIDDEN, HIDDEN, gate);
}
